// coTransformer_53541062312477
// MI455X (gfx1250) — matrix operand order repaired and checked by reading, not compiled, not run on hardware
//
#include <hip/hip_runtime.h>
#include <hip/hip_bf16.h>

// Problem constants
#define MM 4
#define BB 4
#define NN 1024
#define CC 512
#define HH 8
#define DD 64
#define MLPD 4096

typedef __attribute__((ext_vector_type(16))) _Float16 v16h;
typedef __attribute__((ext_vector_type(8)))  float    v8f;

union FragH { v16h h; uint4 u[2]; };
union PackH { _Float16 h[2]; unsigned int u; };

static __device__ __forceinline__ v8f wmma16(v16h a, v16h b, v8f c) {
  // D = A(16x32 f16) * B(32x16 f16) + C(16x16 f32)
  return __builtin_amdgcn_wmma_f32_16x16x32_f16(false, a, false, b, (short)0, c,
                                                false, false);
}
static __device__ __forceinline__ v8f zero8() {
  v8f z = {0.f,0.f,0.f,0.f,0.f,0.f,0.f,0.f};
  return z;
}

// --- DPP-based 16-lane butterfly reductions (no LDS traffic, unlike shfl) ---
// xor-masks {1,2,7,15} span GF(2)^4 -> full reduction within each 16-lane row.
template<int CTRL>
static __device__ __forceinline__ float dppf(float x) {
  int i = __builtin_bit_cast(int, x);
  int r = __builtin_amdgcn_mov_dpp(i, CTRL, 0xf, 0xf, true);
  return __builtin_bit_cast(float, r);
}
static __device__ __forceinline__ float red16_max(float x) {
  x = fmaxf(x, dppf<0xB1>(x));    // quad_perm [1,0,3,2]   (xor 1)
  x = fmaxf(x, dppf<0x4E>(x));    // quad_perm [2,3,0,1]   (xor 2)
  x = fmaxf(x, dppf<0x141>(x));   // row_half_mirror       (xor 7)
  x = fmaxf(x, dppf<0x140>(x));   // row_mirror            (xor 15)
  return x;
}
static __device__ __forceinline__ float red16_sum(float x) {
  x += dppf<0xB1>(x);
  x += dppf<0x4E>(x);
  x += dppf<0x141>(x);
  x += dppf<0x140>(x);
  return x;
}

// ---------------------------------------------------------------------------
// fp32 -> fp16 elementwise convert
// ---------------------------------------------------------------------------
__global__ void cvt_f32_to_f16(const float* __restrict__ src,
                               _Float16* __restrict__ dst, int n) {
  int i = blockIdx.x * blockDim.x + threadIdx.x;
  if (i < n) dst[i] = (_Float16)src[i];
}

// ---------------------------------------------------------------------------
// LayerNorm over C=512, one wave per row, output fp16
// ---------------------------------------------------------------------------
__global__ __launch_bounds__(256)
void ln_f16(const float* __restrict__ x, const float* __restrict__ w,
            const float* __restrict__ b, _Float16* __restrict__ out) {
  const int row  = blockIdx.x * 8 + (threadIdx.x >> 5);
  const int lane = threadIdx.x & 31;
  const int mmod = row >> 12;            // row / (B*N)
  const float* xr = x + (size_t)row * CC;
  float vals[16];
  float s = 0.f, s2 = 0.f;
#pragma unroll
  for (int i = 0; i < 16; ++i) {
    float v = xr[lane + i * 32];
    vals[i] = v; s += v; s2 += v * v;
  }
  s  = red16_sum(s);  s  += __shfl_xor(s,  16, 32);   // cross-half combine
  s2 = red16_sum(s2); s2 += __shfl_xor(s2, 16, 32);
  const float mu   = s  * (1.0f / CC);
  const float var  = s2 * (1.0f / CC) - mu * mu;      // biased variance
  const float rstd = rsqrtf(var + 1e-5f);
  _Float16* orow = out + (size_t)row * CC;
#pragma unroll
  for (int i = 0; i < 16; ++i) {
    int c = lane + i * 32;
    orow[c] = (_Float16)((vals[i] - mu) * rstd * w[mmod * CC + c] + b[mmod * CC + c]);
  }
}

// ---------------------------------------------------------------------------
// Generic tiled WMMA GEMM: Out = A(f16, RxK) * Bw(f16, KxNcols)
// Block tile 64x64, K step 32, 128 threads = 4 waves (wave: 16 rows x 64 cols)
// EPI 0: store f16              (QKV)
// EPI 1: store f16 gelu(acc+b)  (FFN up)
// EPI 2: f32 Out += acc + bias*biasScale   (proj / FFN down, residual)
// ---------------------------------------------------------------------------
template<int EPI>
__global__ __launch_bounds__(128)
void gemm_wmma(const _Float16* __restrict__ A, int lda, long long aZ,
               const _Float16* __restrict__ Bw, int ldb, long long bZ,
               const float* __restrict__ bias, float biasScale, int biasZ,
               void* __restrict__ OutV, int ldo, long long oZ, int K)
{
  __shared__ __align__(16) _Float16 sA [64][40];   // row-major, K padded
  __shared__ __align__(16) _Float16 sBt[64][40];   // transposed: [n][k]

  const int tid  = threadIdx.x;
  const int wave = tid >> 5;
  const int lane = tid & 31;
  const int l0   = lane & 15;
  const int hi   = lane >> 4;
  const int z    = blockIdx.z;

  const _Float16* Ab = A  + (long long)z * aZ + (size_t)(blockIdx.x * 64) * lda;
  const _Float16* Bb = Bw + (long long)z * bZ + blockIdx.y * 64;

  v8f acc[4];
#pragma unroll
  for (int i = 0; i < 4; ++i) acc[i] = zero8();

  const int arow = tid >> 1, aseg = tid & 1;   // A stage: 64 rows x 32 halves
  const int bkp  = tid & 15, bns  = tid >> 4;  // B stage: 16 k-pairs x 8 n-segs

  for (int k0 = 0; k0 < K; k0 += 32) {
    __syncthreads();
    { // stage A tile (row-major), contiguous b128 stores
      const uint4* g = (const uint4*)(Ab + (size_t)arow * lda + k0 + aseg * 16);
      uint4 u0 = g[0], u1 = g[1];
      uint4* s = (uint4*)&sA[arow][aseg * 16];
      s[0] = u0; s[1] = u1;
    }
    { // stage B tile transposed: 2 adjacent k-rows x 8 n -> 8 packed b32 stores
      const int k0g = k0 + 2 * bkp;
      union { uint4 u; _Float16 e[8]; } r0, r1;
      r0.u = *(const uint4*)(Bb + (size_t)(k0g + 0) * ldb + bns * 8);
      r1.u = *(const uint4*)(Bb + (size_t)(k0g + 1) * ldb + bns * 8);
#pragma unroll
      for (int j = 0; j < 8; ++j) {
        PackH p; p.h[0] = r0.e[j]; p.h[1] = r1.e[j];
        *(unsigned int*)&sBt[bns * 8 + j][2 * bkp] = p.u;
      }
    }
    if (k0 + 32 < K) {  // gfx1250 global_prefetch of next tiles
      __builtin_prefetch(Ab + (size_t)arow * lda + k0 + 32 + aseg * 16, 0, 3);
      __builtin_prefetch(Bb + (size_t)(k0 + 32 + 2 * bkp) * ldb + bns * 8, 0, 3);
    }
    __syncthreads();

    // A fragment (ISA 16-bit A layout: e<8 -> k=8*hi+e ; e>=8 -> k=16+8*hi+(e-8))
    FragH fa;
    fa.u[0] = *(const uint4*)&sA[wave * 16 + l0][8 * hi];
    fa.u[1] = *(const uint4*)&sA[wave * 16 + l0][16 + 8 * hi];
#pragma unroll
    for (int nt = 0; nt < 4; ++nt) {
      // B fragment (lane n = l0, e<8 -> k=8*hi+e ; e>=8 -> k=16+8*hi+(e-8))
      FragH fb;
      fb.u[0] = *(const uint4*)&sBt[nt * 16 + l0][8 * hi];
      fb.u[1] = *(const uint4*)&sBt[nt * 16 + l0][16 + 8 * hi];
      acc[nt] = wmma16(fa.h, fb.h, acc[nt]);
    }
  }

  // Epilogue. C/D layout: lane l0+16*hi, vgpr r -> (row = r+8*hi, col = l0)
  const int rowB = blockIdx.x * 64 + wave * 16 + 8 * hi;
  const int colB = blockIdx.y * 64;
  if (EPI == 0 || EPI == 1) {
    _Float16* O = (_Float16*)OutV + (long long)z * oZ;
#pragma unroll
    for (int nt = 0; nt < 4; ++nt) {
      const int col = colB + nt * 16 + l0;
      const float bv = (EPI == 1) ? bias[(long long)z * biasZ + col] : 0.f;
#pragma unroll
      for (int r = 0; r < 8; ++r) {
        float v = acc[nt][r] + bv;
        if (EPI == 1) v = 0.5f * v * (1.0f + erff(v * 0.70710678118f)); // exact GELU
        O[(size_t)(rowB + r) * ldo + col] = (_Float16)v;
      }
    }
  } else {
    float* O = (float*)OutV + (long long)z * oZ;
#pragma unroll
    for (int nt = 0; nt < 4; ++nt) {
      const int col = colB + nt * 16 + l0;
      const float bv = bias[(long long)z * biasZ + col] * biasScale;
#pragma unroll
      for (int r = 0; r < 8; ++r) {
        const size_t idx = (size_t)(rowB + r) * ldo + col;
        O[idx] += acc[nt][r] + bv;
      }
    }
  }
}

// ---------------------------------------------------------------------------
// Flash attention, one (m,b,h) head per blockIdx.{z,y}; 64 queries per block
// (4 waves x 16 queries). K/V streamed through LDS in 32-key chunks.
// qkv buffers f16 [Z=M*B, N, 3*C], q/k/v at +0/+512/+1024, head h at h*64.
// ctx (f32 [M*B, N, C]) is ACCUMULATED (+=) -> zero before first launch.
// joff: kv modality = (q modality + joff) % 4  (0 for self-attention).
// Scale D^-0.5 = 0.125 is folded into the f16 Q fragment (exact, power of 2).
// ---------------------------------------------------------------------------
__global__ __launch_bounds__(128)
void attn_flash(const _Float16* __restrict__ qkvQ,
                const _Float16* __restrict__ qkvKV,
                int joff, float* __restrict__ ctx)
{
  __shared__ __align__(16) _Float16 sK [32][72];      // [key][d]
  __shared__ __align__(16) _Float16 sVt[64][40];      // [d][key]
  __shared__ __align__(16) _Float16 sP [4][16][40];   // per-wave P tile [q][key]

  const int tid  = threadIdx.x;
  const int wave = tid >> 5;
  const int lane = tid & 31;
  const int l0   = lane & 15;
  const int hi   = lane >> 4;

  const int h  = blockIdx.y;
  const int zq = blockIdx.z;                 // m*B + b
  const int m  = zq >> 2, bb = zq & 3;
  const int zkv = (((m + joff) & 3) << 2) | bb;

  const size_t qRow0  = (size_t)zq  * NN;
  const size_t kvRow0 = (size_t)zkv * NN;
  const int qBase = blockIdx.x * 64 + wave * 16;

  // Preload Q fragments (2 halves of D=64), ISA A layout; fold in softmax scale
  FragH qa[2];
  {
    const _Float16* qrow = qkvQ + (qRow0 + qBase + l0) * (3 * CC) + h * DD;
#pragma unroll
    for (int dh = 0; dh < 2; ++dh) {
      qa[dh].u[0] = *(const uint4*)(qrow + dh * 32 + 8 * hi);
      qa[dh].u[1] = *(const uint4*)(qrow + dh * 32 + 16 + 8 * hi);
      qa[dh].h    = qa[dh].h * (_Float16)0.125f;   // D^-0.5, exact
    }
  }

  float rm[8], rl[8];
  v8f o[4];
#pragma unroll
  for (int r = 0; r < 8; ++r) { rm[r] = -1e30f; rl[r] = 0.f; }
#pragma unroll
  for (int t = 0; t < 4; ++t) o[t] = zero8();

  const int skey = tid >> 2, sseg = tid & 3;   // K staging: 32 keys x 4 segs
  const int vkp  = tid >> 3, vds  = tid & 7;   // V staging: 16 key-pairs x 8 d-segs

  for (int kc = 0; kc < NN; kc += 32) {
    __syncthreads();   // all waves done reading previous sK/sVt
    { // stage K (row-major, contiguous b128)
      const _Float16* kg = qkvKV + (kvRow0 + kc + skey) * (3 * CC) + CC + h * DD + sseg * 16;
      uint4 k0v = ((const uint4*)kg)[0], k1v = ((const uint4*)kg)[1];
      uint4* ks = (uint4*)&sK[skey][sseg * 16];
      ks[0] = k0v; ks[1] = k1v;
    }
    { // stage V transposed: 2 adjacent keys x 8 d -> 8 packed b32 stores
      const _Float16* vb0 = qkvKV + (kvRow0 + kc + 2 * vkp)     * (3 * CC) + 2 * CC + h * DD + vds * 8;
      const _Float16* vb1 = qkvKV + (kvRow0 + kc + 2 * vkp + 1) * (3 * CC) + 2 * CC + h * DD + vds * 8;
      union { uint4 u; _Float16 e[8]; } r0, r1;
      r0.u = *(const uint4*)vb0;
      r1.u = *(const uint4*)vb1;
#pragma unroll
      for (int j = 0; j < 8; ++j) {
        PackH p; p.h[0] = r0.e[j]; p.h[1] = r1.e[j];
        *(unsigned int*)&sVt[vds * 8 + j][2 * vkp] = p.u;
      }
    }
    if (kc + 32 < NN) {  // prefetch next chunk's K/V rows
      __builtin_prefetch(qkvKV + (kvRow0 + kc + 32 + skey) * (3 * CC) + CC + h * DD + sseg * 16, 0, 3);
      __builtin_prefetch(qkvKV + (kvRow0 + kc + 32 + 2 * vkp) * (3 * CC) + 2 * CC + h * DD + vds * 8, 0, 3);
    }
    __syncthreads();

    // S = (Q*scale) * K^T  (two 16x16 key tiles)
    v8f s0 = zero8(), s1 = zero8();
#pragma unroll
    for (int dh = 0; dh < 2; ++dh) {
      FragH kb0, kb1;
      kb0.u[0] = *(const uint4*)&sK[l0][dh * 32 + 8 * hi];
      kb0.u[1] = *(const uint4*)&sK[l0][dh * 32 + 16 + 8 * hi];
      kb1.u[0] = *(const uint4*)&sK[16 + l0][dh * 32 + 8 * hi];
      kb1.u[1] = *(const uint4*)&sK[16 + l0][dh * 32 + 16 + 8 * hi];
      s0 = wmma16(qa[dh].h, kb0.h, s0);
      s1 = wmma16(qa[dh].h, kb1.h, s1);
    }

    // Online softmax: DPP butterfly reductions within 16-lane groups
#pragma unroll
    for (int r = 0; r < 8; ++r) {
      const float v0 = s0[r];
      const float v1 = s1[r];
      const float mx = red16_max(fmaxf(v0, v1));
      const float nm   = fmaxf(rm[r], mx);
      const float corr = __expf(rm[r] - nm);
      const float p0   = __expf(v0 - nm);
      const float p1   = __expf(v1 - nm);
      const float sum  = red16_sum(p0 + p1);
      rl[r] = rl[r] * corr + sum;
      rm[r] = nm;
#pragma unroll
      for (int t = 0; t < 4; ++t) o[t][r] *= corr;
      sP[wave][r + 8 * hi][l0]      = (_Float16)p0;
      sP[wave][r + 8 * hi][16 + l0] = (_Float16)p1;
    }
    asm volatile("s_wait_dscnt 0x0" ::: "memory");   // wave-local P visible

    // O += P * V
    FragH pa;
    pa.u[0] = *(const uint4*)&sP[wave][l0][8 * hi];
    pa.u[1] = *(const uint4*)&sP[wave][l0][16 + 8 * hi];
#pragma unroll
    for (int nt = 0; nt < 4; ++nt) {
      FragH vb;
      vb.u[0] = *(const uint4*)&sVt[nt * 16 + l0][8 * hi];
      vb.u[1] = *(const uint4*)&sVt[nt * 16 + l0][16 + 8 * hi];
      o[nt] = wmma16(pa.h, vb.h, o[nt]);
    }
  }

  // Normalize and accumulate into ctx (f32)
#pragma unroll
  for (int r = 0; r < 8; ++r) {
    const float inv = 1.0f / rl[r];
    const int q = qBase + r + 8 * hi;
    float* cp = ctx + ((size_t)zq * NN + q) * CC + h * DD;
#pragma unroll
    for (int nt = 0; nt < 4; ++nt) cp[nt * 16 + l0] += o[nt][r] * inv;
  }
}

// ---------------------------------------------------------------------------
extern "C" void kernel_launch(void* const* d_in, const int* in_sizes, int n_in,
                              void* d_out, int out_size, void* d_ws, size_t ws_size,
                              hipStream_t stream)
{
  (void)in_sizes; (void)n_in; (void)out_size; (void)ws_size;

  const float* sa_ln_w   = (const float*)d_in[4];
  const float* sa_ln_b   = (const float*)d_in[5];
  const float* sa_qkv_w  = (const float*)d_in[6];
  const float* sa_proj_w = (const float*)d_in[7];
  const float* sa_proj_b = (const float*)d_in[8];
  const float* ca_ln_w   = (const float*)d_in[9];
  const float* ca_ln_b   = (const float*)d_in[10];
  const float* ca_qkv_w  = (const float*)d_in[11];
  const float* ca_proj_w = (const float*)d_in[12];
  const float* ca_proj_b = (const float*)d_in[13];
  const float* ffn_ln_w  = (const float*)d_in[14];
  const float* ffn_ln_b  = (const float*)d_in[15];
  const float* ffn_w1    = (const float*)d_in[16];
  const float* ffn_b1    = (const float*)d_in[17];
  const float* ffn_w2    = (const float*)d_in[18];
  const float* ffn_b2    = (const float*)d_in[19];

  float* x = (float*)d_out;                       // residual state [M,B,N,C]
  const size_t xE = (size_t)MM * BB * NN * CC;    // 8388608

  // workspace carving (256B aligned)
  char* wsb = (char*)d_ws;
  size_t off = 0;
  auto alloc = [&](size_t bytes) -> void* {
    void* p = (void*)(wsb + off);
    off += (bytes + 255) & ~(size_t)255;
    return p;
  };
  _Float16* xh     = (_Float16*)alloc(xE * 2);                            // LN out
  _Float16* qkvh   = (_Float16*)alloc((size_t)MM * BB * NN * 3 * CC * 2); // QKV f16
  float*    ctx    = (float*)   alloc(xE * 4);                            // attn ctx f32
  _Float16* ctxh   = (_Float16*)alloc(xE * 2);
  _Float16* wsaqkv = (_Float16*)alloc((size_t)MM * CC * 3 * CC * 2);
  _Float16* wcaqkv = (_Float16*)alloc((size_t)MM * CC * 3 * CC * 2);
  _Float16* wsaprj = (_Float16*)alloc((size_t)MM * CC * CC * 2);
  _Float16* wcaprj = (_Float16*)alloc((size_t)MM * CC * CC * 2);
  _Float16* wffn1  = (_Float16*)alloc((size_t)MM * CC * MLPD * 2);
  _Float16* wffn2  = (_Float16*)alloc((size_t)MM * MLPD * CC * 2);
  _Float16* hbuf   = (_Float16*)alloc((size_t)BB * NN * MLPD * 2);        // per-m FFN hidden

  auto cvt = [&](const float* src, _Float16* dst, size_t n) {
    cvt_f32_to_f16<<<dim3((unsigned)((n + 255) / 256)), 256, 0, stream>>>(src, dst, (int)n);
  };

  // x <- inputs
  for (int i = 0; i < 4; ++i)
    hipMemcpyAsync(x + (size_t)i * BB * NN * CC, d_in[i],
                   (size_t)BB * NN * CC * sizeof(float),
                   hipMemcpyDeviceToDevice, stream);

  // weights -> f16
  cvt(sa_qkv_w,  wsaqkv, (size_t)MM * CC * 3 * CC);
  cvt(ca_qkv_w,  wcaqkv, (size_t)MM * CC * 3 * CC);
  cvt(sa_proj_w, wsaprj, (size_t)MM * CC * CC);
  cvt(ca_proj_w, wcaprj, (size_t)MM * CC * CC);
  cvt(ffn_w1,    wffn1,  (size_t)MM * CC * MLPD);
  cvt(ffn_w2,    wffn2,  (size_t)MM * MLPD * CC);

  const dim3 lnGrid(MM * BB * NN / 8);
  const dim3 qkvGrid(BB * NN / 64, 3 * CC / 64, MM);
  const dim3 prjGrid(BB * NN / 64, CC / 64, MM);
  const dim3 attGrid(NN / 64, HH, MM * BB);

  // ===== Self-attention block =====
  ln_f16<<<lnGrid, 256, 0, stream>>>(x, sa_ln_w, sa_ln_b, xh);
  gemm_wmma<0><<<qkvGrid, 128, 0, stream>>>(
      xh, CC, (long long)BB * NN * CC,
      wsaqkv, 3 * CC, (long long)CC * 3 * CC,
      nullptr, 0.f, 0,
      qkvh, 3 * CC, (long long)BB * NN * 3 * CC, CC);
  hipMemsetAsync(ctx, 0, xE * sizeof(float), stream);
  attn_flash<<<attGrid, 128, 0, stream>>>(qkvh, qkvh, 0, ctx);
  cvt(ctx, ctxh, xE);
  gemm_wmma<2><<<prjGrid, 128, 0, stream>>>(
      ctxh, CC, (long long)BB * NN * CC,
      wsaprj, CC, (long long)CC * CC,
      sa_proj_b, 1.0f, CC,
      x, CC, (long long)BB * NN * CC, CC);

  // ===== Cross-attention block (3 kv modalities; same proj_i => sum contexts) =====
  ln_f16<<<lnGrid, 256, 0, stream>>>(x, ca_ln_w, ca_ln_b, xh);
  gemm_wmma<0><<<qkvGrid, 128, 0, stream>>>(
      xh, CC, (long long)BB * NN * CC,
      wcaqkv, 3 * CC, (long long)CC * 3 * CC,
      nullptr, 0.f, 0,
      qkvh, 3 * CC, (long long)BB * NN * 3 * CC, CC);
  hipMemsetAsync(ctx, 0, xE * sizeof(float), stream);
  for (int joff = 1; joff <= 3; ++joff)
    attn_flash<<<attGrid, 128, 0, stream>>>(qkvh, qkvh, joff, ctx);
  cvt(ctx, ctxh, xE);
  gemm_wmma<2><<<prjGrid, 128, 0, stream>>>(
      ctxh, CC, (long long)BB * NN * CC,
      wcaprj, CC, (long long)CC * CC,
      ca_proj_b, 3.0f, CC,              // (M-1) * bias
      x, CC, (long long)BB * NN * CC, CC);

  // ===== FFN block (per modality to bound workspace) =====
  ln_f16<<<lnGrid, 256, 0, stream>>>(x, ffn_ln_w, ffn_ln_b, xh);
  for (int m = 0; m < MM; ++m) {
    gemm_wmma<1><<<dim3(BB * NN / 64, MLPD / 64, 1), 128, 0, stream>>>(
        xh + (size_t)m * BB * NN * CC, CC, 0,
        wffn1 + (size_t)m * CC * MLPD, MLPD, 0,
        ffn_b1 + (size_t)m * MLPD, 1.0f, 0,
        hbuf, MLPD, 0, CC);
    gemm_wmma<2><<<dim3(BB * NN / 64, CC / 64, 1), 128, 0, stream>>>(
        hbuf, MLPD, 0,
        wffn2 + (size_t)m * MLPD * CC, CC, 0,
        ffn_b2 + (size_t)m * CC, 1.0f, 0,
        x + (size_t)m * BB * NN * CC, CC, 0, MLPD);
  }
}
